// GAT_88545045775103
// MI455X (gfx1250) — compile-verified, matrix operand order checked by reading, not run on hardware
//
#include <hip/hip_runtime.h>
#include <hip/hip_bf16.h>
#include <float.h>

// Problem constants (from reference)
#define NB     2
#define NSEQ   512
#define NODE_H 128
#define EDGE_H 64
#define HDTOT  128   // N_HEAD * HEAD_HIDDEN

typedef float v2f  __attribute__((ext_vector_type(2)));
typedef float v8f  __attribute__((ext_vector_type(8)));
typedef __bf16 v4bf  __attribute__((ext_vector_type(4)));
typedef __bf16 v8bf  __attribute__((ext_vector_type(8)));
typedef __bf16 v16bf __attribute__((ext_vector_type(16)));

// 16-bit A-frag (16x32, M x K): lane(m=l&15, half=l>>4) holds
//   elements [0..7]  = A[m][ks*32 + 8*half + 0..7]
//   elements [8..15] = A[m][ks*32 + 16 + 8*half + 0..7]
// B-frag (32x16, K x N) mirrors with n = l&15; we store B transposed [n][k]
// so both sides load two contiguous 16-byte runs from LDS.
__device__ __forceinline__ v16bf frag16(const __bf16* rowbase, int ks, int half) {
    const __bf16* p = rowbase + ks * 32 + 8 * half;
    v8bf a = *(const v8bf*)p;
    v8bf b = *(const v8bf*)(p + 16);
    return __builtin_shufflevector(a, b, 0,1,2,3,4,5,6,7,8,9,10,11,12,13,14,15);
}

struct bfpair { __bf16 hi, lo; };
__device__ __forceinline__ bfpair split_bf16(float v) {
    bfpair p;
    p.hi = (__bf16)v;                    // RNE, top ~8 mantissa bits
    p.lo = (__bf16)(v - (float)p.hi);    // next ~8 bits; residual ~2^-16 rel
    return p;
}

// -------------------------------------------------------------------------
// Kernel 1: qkv = x @ Wqkv + bqkv (1024 x 384 x 128). Tiny; exact f32 WMMA.
// -------------------------------------------------------------------------
__global__ __launch_bounds__(32) void qkv_kernel(
    const float* __restrict__ x, const float* __restrict__ Wqkv,
    const float* __restrict__ bqkv,
    float* __restrict__ q_ws, float* __restrict__ k_ws, float* __restrict__ v_ws)
{
    const int lane = threadIdx.x;
    const int m16  = lane & 15;
    const int half = lane >> 4;
    const int R0   = blockIdx.x * 16;
    const int ct   = blockIdx.y;

    v8f c = {0.f,0.f,0.f,0.f,0.f,0.f,0.f,0.f};
    const float* arow = x + (size_t)(R0 + m16) * NODE_H;
    const float* bcol = Wqkv + (ct * 16 + m16);
    #pragma unroll
    for (int kb = 0; kb < NODE_H; kb += 4) {
        const int k2 = kb + 2 * half;
        v2f a = *(const v2f*)(arow + k2);
        v2f b;
        b.x = bcol[(size_t)k2 * 384];
        b.y = bcol[(size_t)(k2 + 1) * 384];
        c = __builtin_amdgcn_wmma_f32_16x16x4_f32(false, a, false, b,
                                                  (short)0, c, false, false);
    }
    const int g = ct * 16 + m16;
    const float bias = bqkv[g];
    float* dst; int gc;
    if (g < 128)      { dst = q_ws; gc = g;       }
    else if (g < 256) { dst = k_ws; gc = g - 128; }
    else              { dst = v_ws; gc = g - 256; }
    #pragma unroll
    for (int r = 0; r < 8; ++r)
        dst[(size_t)(R0 + r + 8 * half) * HDTOT + gc] = c[r] + bias;
}

// -------------------------------------------------------------------------
// Kernel 2: fused edge projection + attention + outputs. One workgroup per
// (b,i), 4 waves, wave w owns j-tiles w, w+4, ... (8 tiles each).
// Hot GEMMs run on v_wmma_f32_16x16x32_bf16 with hi/lo split (3 products),
// f32 accumulate -> ~2^-16 relative error vs the f32 reference.
// -------------------------------------------------------------------------
__global__ __launch_bounds__(128) void fused_kernel(
    const float* __restrict__ edge_attr, const int* __restrict__ mask,
    const float* __restrict__ We,  const float* __restrict__ be,
    const float* __restrict__ Wno, const float* __restrict__ bno,
    const float* __restrict__ Weo, const float* __restrict__ beo,
    const float* __restrict__ q_ws, const float* __restrict__ k_ws,
    const float* __restrict__ v_ws,
    float* __restrict__ node_out, float* __restrict__ edge_out)
{
    // bf16 rows padded to 72 / 136 elements: fragment B128 reads hit each
    // bank pair at most twice (optimal for 512B/wave loads).
    __shared__ __align__(16) __bf16 sWeHi[128 * 72],  sWeLo[128 * 72];   // We^T [col][k<64]
    __shared__ __align__(16) __bf16 sWeoHi[64 * 136], sWeoLo[64 * 136];  // Weo^T[col][k<128]
    __shared__ __align__(16) __bf16 sOnes[16 * 136];                     // head-indicator^T
    __shared__ __align__(16) __bf16 sAHi[4][16 * 72],  sALo[4][16 * 72]; // EA tile hi/lo
    __shared__ __align__(16) __bf16 sRHi[4][16 * 136], sRLo[4][16 * 136];// raw tile hi/lo
    __shared__ __align__(16) float  sEdge[4][16 * 68];                   // edge staging
    __shared__ float sS[4][128];                    // probs [j*8+h]
    __shared__ float sM[4][8], sL[4][8], sScale[4][8];
    __shared__ float sQ[128], sBe[128], sBeo[64], sNode[128];

    const int tid  = threadIdx.x;
    const int w    = tid >> 5, lane = tid & 31;
    const int m16  = lane & 15, half = lane >> 4;
    const int b    = blockIdx.x >> 9, i = blockIdx.x & 511;
    const size_t bi = (size_t)(b * NSEQ + i);

    // ---- init LDS: split weights into bf16 hi/lo, build ones matrix ----
    for (int f = tid; f < 64 * 128; f += 128) {        // We (64x128)
        const int k = f >> 7, cc = f & 127;
        const bfpair p = split_bf16(We[f]);
        sWeHi[cc * 72 + k] = p.hi; sWeLo[cc * 72 + k] = p.lo;
    }
    for (int f = tid; f < 128 * 64; f += 128) {        // Weo (128x64)
        const int k = f >> 6, n = f & 63;
        const bfpair p = split_bf16(Weo[f]);
        sWeoHi[n * 136 + k] = p.hi; sWeoLo[n * 136 + k] = p.lo;
    }
    for (int f = tid; f < 16 * 136; f += 128) {        // Ones^T: [h][k]=(k>>4==h)
        const int h = f / 136, k = f - h * 136;
        sOnes[f] = (k < 128 && (k >> 4) == h) ? (__bf16)1.0f : (__bf16)0.0f;
    }
    sQ[tid]  = q_ws[bi * HDTOT + tid];
    sBe[tid] = be[tid];
    if (tid < 64) sBeo[tid] = beo[tid];
    if (tid < 32) { sM[tid >> 3][tid & 7] = -FLT_MAX; sL[tid >> 3][tid & 7] = 0.f; }
    __syncthreads();

    float qreg[8];
    #pragma unroll
    for (int ct = 0; ct < 8; ++ct) qreg[ct] = sQ[ct * 16 + m16];

    float acc[4] = {0.f, 0.f, 0.f, 0.f};               // node accum, f = q4*32+lane

    const float* ea_base = edge_attr + (bi * NSEQ) * EDGE_H;
    float*       eo_base = edge_out  + (bi * NSEQ) * EDGE_H;

    for (int jt = w; jt < NSEQ / 16; jt += 4) {
        const int j0 = jt * 16;

        // prefetch next tile this wave will touch (one 128B line per lane)
        if (jt + 4 < NSEQ / 16)
            __builtin_prefetch(ea_base + (size_t)(j0 + 64) * EDGE_H + lane * 32, 0, 1);

        // (1) load contiguous 16x64 f32 edge_attr tile, split to bf16 hi/lo
        const float4* src = (const float4*)(ea_base + (size_t)j0 * EDGE_H);
        #pragma unroll
        for (int it = 0; it < 8; ++it) {
            const int idx = it * 32 + lane;
            const int r = idx >> 4, c4 = idx & 15;
            const float4 vv = src[idx];
            const bfpair p0 = split_bf16(vv.x), p1 = split_bf16(vv.y);
            const bfpair p2 = split_bf16(vv.z), p3 = split_bf16(vv.w);
            v4bf h, l;
            h[0] = p0.hi; h[1] = p1.hi; h[2] = p2.hi; h[3] = p3.hi;
            l[0] = p0.lo; l[1] = p1.lo; l[2] = p2.lo; l[3] = p3.lo;
            *(v4bf*)&sAHi[w][r * 72 + c4 * 4] = h;
            *(v4bf*)&sALo[w][r * 72 + c4 * 4] = l;
        }
        __syncthreads();

        // (2) E = EA @ We (+be), fused raw = q*k*E -> bf16 hi/lo raw tiles
        v16bf aH0 = frag16(&sAHi[w][m16 * 72], 0, half);
        v16bf aL0 = frag16(&sALo[w][m16 * 72], 0, half);
        v16bf aH1 = frag16(&sAHi[w][m16 * 72], 1, half);
        v16bf aL1 = frag16(&sALo[w][m16 * 72], 1, half);
        for (int ct = 0; ct < 8; ++ct) {
            const int col = ct * 16 + m16;
            v8f c = {0,0,0,0,0,0,0,0};
            {
                v16bf bH = frag16(&sWeHi[col * 72], 0, half);
                v16bf bL = frag16(&sWeLo[col * 72], 0, half);
                c = __builtin_amdgcn_wmma_f32_16x16x32_bf16(false, aH0, false, bH, (short)0, c, false, false);
                c = __builtin_amdgcn_wmma_f32_16x16x32_bf16(false, aH0, false, bL, (short)0, c, false, false);
                c = __builtin_amdgcn_wmma_f32_16x16x32_bf16(false, aL0, false, bH, (short)0, c, false, false);
            }
            {
                v16bf bH = frag16(&sWeHi[col * 72], 1, half);
                v16bf bL = frag16(&sWeLo[col * 72], 1, half);
                c = __builtin_amdgcn_wmma_f32_16x16x32_bf16(false, aH1, false, bH, (short)0, c, false, false);
                c = __builtin_amdgcn_wmma_f32_16x16x32_bf16(false, aH1, false, bL, (short)0, c, false, false);
                c = __builtin_amdgcn_wmma_f32_16x16x32_bf16(false, aL1, false, bH, (short)0, c, false, false);
            }
            const float qv = qreg[ct], bev = sBe[col];
            #pragma unroll
            for (int r = 0; r < 8; ++r) {
                const int row = r + 8 * half;
                const float kv = k_ws[((size_t)b * NSEQ + j0 + row) * HDTOT + col];
                const float rv = qv * kv * (c[r] + bev);   // raw (unmasked)
                const bfpair pr = split_bf16(rv);
                sRHi[w][row * 136 + col] = pr.hi;
                sRLo[w][row * 136 + col] = pr.lo;
            }
        }
        __syncthreads();

        // (3) logits via WMMA: s[j][h] = raw(16x128) @ Ones(128x8) / 4, masked
        {
            v8f c = {0,0,0,0,0,0,0,0};
            #pragma unroll
            for (int ks = 0; ks < 4; ++ks) {
                v16bf aH = frag16(&sRHi[w][m16 * 136], ks, half);
                v16bf aL = frag16(&sRLo[w][m16 * 136], ks, half);
                v16bf bO = frag16(&sOnes[m16 * 136],   ks, half);
                c = __builtin_amdgcn_wmma_f32_16x16x32_bf16(false, aH, false, bO, (short)0, c, false, false);
                c = __builtin_amdgcn_wmma_f32_16x16x32_bf16(false, aL, false, bO, (short)0, c, false, false);
            }
            if (m16 < 8) {
                const int h = m16;
                #pragma unroll
                for (int r = 0; r < 8; ++r) {
                    const int j = r + 8 * half;
                    const bool mk = mask[b * NSEQ + j0 + j] != 0;
                    sS[w][j * 8 + h] = mk ? c[r] * 0.25f : -FLT_MAX;
                }
            }
        }
        __syncthreads();

        // (4) per-head online-softmax update (lanes 0..7 of each wave)
        if (lane < 8) {
            const int h = lane;
            const float mo = sM[w][h];
            float tm = -FLT_MAX;
            for (int j = 0; j < 16; ++j) tm = fmaxf(tm, sS[w][j * 8 + h]);
            const float mn = fmaxf(mo, tm);
            const float sc = __expf(mo - mn);
            float l = sL[w][h] * sc;
            for (int j = 0; j < 16; ++j) {
                const float p = __expf(sS[w][j * 8 + h] - mn);
                sS[w][j * 8 + h] = p;
                l += p;
            }
            sM[w][h] = mn; sL[w][h] = l; sScale[w][h] = sc;
        }
        __syncthreads();

        // (5) node accumulator: acc[h,d] = acc*scale + sum_j p[j,h]*v[j,h,d]
        #pragma unroll
        for (int q4 = 0; q4 < 4; ++q4) {
            const int f = q4 * 32 + lane;
            const int h = f >> 4;
            float a = acc[q4] * sScale[w][h];
            for (int j = 0; j < 16; ++j)
                a += sS[w][j * 8 + h] * v_ws[((size_t)b * NSEQ + j0 + j) * HDTOT + f];
            acc[q4] = a;
        }

        // (6) edge_out tile = raw(16x128) @ Weo(128x64) + beo
        for (int ct = 0; ct < 4; ++ct) {
            const int col = ct * 16 + m16;
            v8f c = {0,0,0,0,0,0,0,0};
            #pragma unroll
            for (int ks = 0; ks < 4; ++ks) {
                v16bf aH = frag16(&sRHi[w][m16 * 136], ks, half);
                v16bf aL = frag16(&sRLo[w][m16 * 136], ks, half);
                v16bf bH = frag16(&sWeoHi[col * 136],  ks, half);
                v16bf bL = frag16(&sWeoLo[col * 136],  ks, half);
                c = __builtin_amdgcn_wmma_f32_16x16x32_bf16(false, aH, false, bH, (short)0, c, false, false);
                c = __builtin_amdgcn_wmma_f32_16x16x32_bf16(false, aH, false, bL, (short)0, c, false, false);
                c = __builtin_amdgcn_wmma_f32_16x16x32_bf16(false, aL, false, bH, (short)0, c, false, false);
            }
            const float bev = sBeo[col];
            #pragma unroll
            for (int r = 0; r < 8; ++r)
                sEdge[w][(r + 8 * half) * 68 + col] = c[r] + bev;
        }
        __syncthreads();

        // (7) coalesced float4 store of the contiguous 16x64 edge_out tile
        float4* dst = (float4*)(eo_base + (size_t)j0 * EDGE_H);
        #pragma unroll
        for (int it = 0; it < 8; ++it) {
            const int idx = it * 32 + lane;
            dst[idx] = *(const float4*)&sEdge[w][(idx >> 4) * 68 + (idx & 15) * 4];
        }
        __syncthreads();
    }

    // ---- merge the 4 waves' online-softmax states, then node_out ----
    #pragma unroll
    for (int q4 = 0; q4 < 4; ++q4) sEdge[w][q4 * 32 + lane] = acc[q4];
    __syncthreads();
    {
        const int f = tid, h = f >> 4;
        const float M = fmaxf(fmaxf(sM[0][h], sM[1][h]), fmaxf(sM[2][h], sM[3][h]));
        float L = 0.f, nv = 0.f;
        #pragma unroll
        for (int w2 = 0; w2 < 4; ++w2) {
            const float e = __expf(sM[w2][h] - M);
            L  += sL[w2][h] * e;
            nv += sEdge[w2][f] * e;
        }
        sNode[f] = nv / L;
    }
    __syncthreads();
    {
        float o = bno[tid];
        for (int k = 0; k < HDTOT; ++k)
            o += sNode[k] * Wno[(size_t)k * NODE_H + tid];   // coalesced over tid
        node_out[bi * NODE_H + tid] = o;
    }
}

// -------------------------------------------------------------------------
extern "C" void kernel_launch(void* const* d_in, const int* in_sizes, int n_in,
                              void* d_out, int out_size, void* d_ws, size_t ws_size,
                              hipStream_t stream)
{
    (void)in_sizes; (void)n_in; (void)out_size; (void)ws_size;
    const float* x         = (const float*)d_in[0];
    const float* edge_attr = (const float*)d_in[1];
    const int*   mask      = (const int*)  d_in[2];
    const float* Wqkv      = (const float*)d_in[3];
    const float* bqkv      = (const float*)d_in[4];
    const float* We        = (const float*)d_in[5];
    const float* be        = (const float*)d_in[6];
    const float* Wno       = (const float*)d_in[7];
    const float* bno       = (const float*)d_in[8];
    const float* Weo       = (const float*)d_in[9];
    const float* beo       = (const float*)d_in[10];

    float* q_ws = (float*)d_ws;                       // 3 * 2*512*128 f32 = 1.5 MB
    float* k_ws = q_ws + (size_t)NB * NSEQ * HDTOT;
    float* v_ws = k_ws + (size_t)NB * NSEQ * HDTOT;

    float* node_out = (float*)d_out;                              // 2*512*128
    float* edge_out = node_out + (size_t)NB * NSEQ * NODE_H;      // 2*512*512*64

    dim3 g1((NB * NSEQ) / 16, 384 / 16);
    qkv_kernel<<<g1, 32, 0, stream>>>(x, Wqkv, bqkv, q_ws, k_ws, v_ws);
    fused_kernel<<<NB * NSEQ, 128, 0, stream>>>(edge_attr, mask, We, be, Wno, bno,
                                                Weo, beo, q_ws, k_ws, v_ws,
                                                node_out, edge_out);
}
